// CausalMultiheadAttn_62689342652862
// MI455X (gfx1250) — matrix operand order repaired and checked by reading, not compiled, not run on hardware
//
#include <hip/hip_runtime.h>
#include <hip/hip_bf16.h>
#include <math.h>

typedef __attribute__((ext_vector_type(16))) __bf16 v16bf;
typedef __attribute__((ext_vector_type(8)))  float  v8f;

#define D_MODEL      2048
#define NUM_HEADS    16
#define NUM_KV_HEADS 4
#define D_HEAD       128
#define GROUPS       4
#define SEQ          2048
#define BATCH        2

// A-matrix (16x32 bf16) per-lane element -> K index (ISA 7.12.2)
__device__ __forceinline__ int a_kidx(int e, int lane) {
  int p = e >> 1;
  return ((p & 4) << 2) | ((lane >> 4) << 3) | ((p & 3) << 1) | (e & 1);
}
// B-matrix (32x16 bf16) per-lane element -> K index
__device__ __forceinline__ int b_kidx(int e, int lane) {
  return ((e & 8) << 1) | ((lane >> 4) << 3) | (e & 7);
}

// CDNA5 async copy: global -> LDS, 16 bytes, tracked by ASYNCcnt.
// (generic pointers to __shared__ carry the LDS byte offset in their low 32 bits)
__device__ __forceinline__ void async_b128(void* lds_dst, const void* gsrc) {
  asm volatile("global_load_async_to_lds_b128 %0, %1, off"
               :: "v"((unsigned)(size_t)lds_dst),
                  "v"((unsigned long long)(size_t)gsrc)
               : "memory");
}
__device__ __forceinline__ void wait_async0() {
  asm volatile("s_wait_asynccnt 0x0" ::: "memory");
}

// ---------------------------------------------------------------------------
// elementwise f32 -> bf16 (one-time precision shift; bandwidth-trivial)
// ---------------------------------------------------------------------------
__global__ __launch_bounds__(256) void cvt_f32_bf16(
    const float* __restrict__ in, __bf16* __restrict__ out, int n)
{
  int i = (blockIdx.x * 256 + threadIdx.x) * 8;
  if (i >= n) return;
  float4 a = *(const float4*)(in + i);
  float4 b = *(const float4*)(in + i + 4);
  union { __bf16 h[8]; uint4 u; } pk;
  pk.h[0] = (__bf16)a.x; pk.h[1] = (__bf16)a.y; pk.h[2] = (__bf16)a.z; pk.h[3] = (__bf16)a.w;
  pk.h[4] = (__bf16)b.x; pk.h[5] = (__bf16)b.y; pk.h[6] = (__bf16)b.z; pk.h[7] = (__bf16)b.w;
  *(uint4*)(out + i) = pk.u;
}

// ---------------------------------------------------------------------------
// Y[M,N] = X[M,K] * W[N,K]^T  (bf16 in, f32 out)
// 128 threads = 4 waves; block tile 128x128; wave tile 64x64
// => 16 WMMAs per K-step from 8 fragment loads (high register reuse).
// Double-buffered LDS filled by async b128 copies (ASYNCcnt pipelined).
// ---------------------------------------------------------------------------
__global__ __launch_bounds__(128) void gemm_bf16(
    const __bf16* __restrict__ X, const __bf16* __restrict__ W,
    float* __restrict__ Y, int M, int N, int K)
{
  __shared__ __align__(16) __bf16 As[2][128][40];  // +8 pad, 80B row stride (16B mult)
  __shared__ __align__(16) __bf16 Bs[2][128][40];

  const int tid  = threadIdx.x;
  const int lane = tid & 31;
  const int wave = tid >> 5;
  const int wm   = (wave & 1) * 64;     // 2 row groups of 64
  const int wn   = (wave >> 1) * 64;    // 2 col groups of 64
  const int ln   = lane & 15;
  const int half = lane >> 4;
  const int mBase = blockIdx.y * 128;
  const int nBase = blockIdx.x * 128;

  v8f acc[4][4];
  const v8f vzero = {};
  #pragma unroll
  for (int i = 0; i < 4; ++i)
    #pragma unroll
    for (int j = 0; j < 4; ++j) acc[i][j] = vzero;

  const int nt = K / 32;
  // tile issue: A and B are each 512 b128 chunks (128 rows x 4 chunks)
  auto issue = [&](int buf, int kk) {
    const __bf16* xs = X + (size_t)mBase * K + kk;
    const __bf16* wsrc = W + (size_t)nBase * K + kk;
    #pragma unroll
    for (int i = 0; i < 4; ++i) {
      int c = tid + i * 128;
      int row = c >> 2, col = (c & 3) << 3;
      async_b128(&As[buf][row][col], xs + (size_t)row * K + col);
      async_b128(&Bs[buf][row][col], wsrc + (size_t)row * K + col);
    }
  };

  issue(0, 0);
  wait_async0();
  __syncthreads();

  int buf = 0;
  for (int t = 0; t < nt; ++t) {
    if (t + 1 < nt) issue(buf ^ 1, (t + 1) * 32);   // overlap copy with WMMA

    v16bf a[4], bb[4];
    #pragma unroll
    for (int e = 0; e < 16; ++e) {
      int ka = a_kidx(e, lane), kb = b_kidx(e, lane);
      #pragma unroll
      for (int i = 0; i < 4; ++i) {
        a[i][e]  = As[buf][wm + i * 16 + ln][ka];
        bb[i][e] = Bs[buf][wn + i * 16 + ln][kb];
      }
    }
    #pragma unroll
    for (int i = 0; i < 4; ++i)
      #pragma unroll
      for (int j = 0; j < 4; ++j)
        acc[i][j] = __builtin_amdgcn_wmma_f32_16x16x32_bf16(
            false, a[i], false, bb[j], (short)0, acc[i][j], false, false);

    if (t + 1 < nt) { wait_async0(); __syncthreads(); }
    buf ^= 1;
  }

  #pragma unroll
  for (int i = 0; i < 4; ++i)
    #pragma unroll
    for (int j = 0; j < 4; ++j)
      #pragma unroll
      for (int r = 0; r < 8; ++r) {
        int row = mBase + wm + i * 16 + r + (half << 3);
        int col = nBase + wn + j * 16 + ln;
        Y[(size_t)row * N + col] = acc[i][j][r];
      }
}

// ---------------------------------------------------------------------------
// RMSNorm + RoPE (+ head-major transpose), f32 -> bf16
// in: [B*S, nheads, 128]  out: [B, nheads, S, 128]
// ---------------------------------------------------------------------------
__global__ __launch_bounds__(128) void norm_rope_kernel(
    const float* __restrict__ in, const float* __restrict__ gamma,
    __bf16* __restrict__ out, int nheads, int S, int doRope)
{
  const int lane = threadIdx.x & 31;
  const int wave = threadIdx.x >> 5;
  const int bs   = blockIdx.x;
  const int s    = bs % S;
  const int b    = bs / S;
  const int h    = blockIdx.y * 4 + wave;
  const int d0   = lane * 4;

  float4 xv = *(const float4*)(in + ((size_t)bs * nheads + h) * D_HEAD + d0);
  float x[4] = {xv.x, xv.y, xv.z, xv.w};

  if (doRope) {
    float ss = x[0]*x[0] + x[1]*x[1] + x[2]*x[2] + x[3]*x[3];
    #pragma unroll
    for (int off = 1; off < 32; off <<= 1) ss += __shfl_xor(ss, off, 32);
    float rinv = rsqrtf(ss * (1.0f / D_HEAD) + 1e-6f);
    #pragma unroll
    for (int j = 0; j < 4; ++j) x[j] = x[j] * rinv * gamma[d0 + j];
    #pragma unroll
    for (int j = 0; j < 4; ++j) {
      float other = __shfl_xor(x[j], 16, 32);          // partner at d ^ 64
      float rot   = (lane < 16) ? -other : other;      // [-x2, x1]
      int   idx   = (d0 + j) & 63;
      float invf  = __expf((float)idx * (-9.210340371976184f / 64.0f));
      float ang   = (float)s * invf;
      x[j] = x[j] * __cosf(ang) + rot * __sinf(ang);
    }
  }
  __bf16* dst = out + (((size_t)b * nheads + h) * S + s) * D_HEAD + d0;
  #pragma unroll
  for (int j = 0; j < 4; ++j) dst[j] = (__bf16)x[j];
}

// ---------------------------------------------------------------------------
// Causal GQA flash attention; K/V tiles double-buffered via async-to-LDS.
// block: 128 threads = 4 waves; block owns 64 q rows, wave owns 16.
// out: bf16 [B, S, H*128] (feeds output-projection GEMM directly)
// ---------------------------------------------------------------------------
__global__ __launch_bounds__(128) void flash_attn_kernel(
    const __bf16* __restrict__ Qh, const __bf16* __restrict__ Kh,
    const __bf16* __restrict__ Vh, __bf16* __restrict__ att, int S)
{
  __shared__ __align__(16) __bf16 Ks[2][32][136];  // 272B row stride (16B mult)
  __shared__ __align__(16) __bf16 Vs[2][32][136];
  __shared__ __align__(16) __bf16 Ps[4][512];      // per-wave P staging 16x32

  const int tid  = threadIdx.x;
  const int lane = tid & 31;
  const int wave = tid >> 5;
  const int half = lane >> 4;
  const int ln   = lane & 15;

  const int qblk  = blockIdx.x;
  const int h     = blockIdx.y;
  const int b     = blockIdx.z;
  const int kvh   = h / GROUPS;
  const int qrow0 = qblk * 64 + wave * 16;

  const __bf16* ksrc0 = Kh + ((size_t)b * NUM_KV_HEADS + kvh) * S * D_HEAD;
  const __bf16* vsrc0 = Vh + ((size_t)b * NUM_KV_HEADS + kvh) * S * D_HEAD;

  auto issue_kv = [&](int buf, int kvbase) {
    const __bf16* ks = ksrc0 + (size_t)kvbase * D_HEAD;
    const __bf16* vs = vsrc0 + (size_t)kvbase * D_HEAD;
    #pragma unroll
    for (int i = 0; i < 4; ++i) {              // 512 chunks each of K and V
      int c = tid + i * 128;
      int row = c >> 4, col = (c & 15) << 3;
      async_b128(&Ks[buf][row][col], ks + (size_t)row * D_HEAD + col);
      async_b128(&Vs[buf][row][col], vs + (size_t)row * D_HEAD + col);
    }
  };

  // Q fragments: 16 rows x 128 as 4 A-fragments (registers)
  v16bf qf[4];
  {
    const __bf16* qbase = Qh + (((size_t)b * NUM_HEADS + h) * S + qrow0 + ln) * D_HEAD;
    #pragma unroll
    for (int t = 0; t < 4; ++t)
      #pragma unroll
      for (int e = 0; e < 16; ++e)
        qf[t][e] = qbase[t * 32 + a_kidx(e, lane)];
  }

  const v8f vzero = {};
  v8f o[8];
  #pragma unroll
  for (int g = 0; g < 8; ++g) o[g] = vzero;
  float lsum[8], mrun[8];
  #pragma unroll
  for (int r = 0; r < 8; ++r) { lsum[r] = 0.0f; mrun[r] = -1e30f; }

  const float scale = 0.08838834764831845f;   // 1/sqrt(128)
  const int nj = qblk * 2 + 2;                // causal block bound

  issue_kv(0, 0);
  wait_async0();
  __syncthreads();

  int buf = 0;
  for (int j = 0; j < nj; ++j) {
    const int kvbase = j * 32;
    if (j + 1 < nj) issue_kv(buf ^ 1, kvbase + 32);   // overlap copy with math

    // scores S = Q K^T : 16x32
    v8f s0 = vzero, s1 = vzero;
    #pragma unroll
    for (int t = 0; t < 4; ++t) {
      v16bf b0, b1;
      #pragma unroll
      for (int e = 0; e < 16; ++e) {
        int kc = t * 32 + b_kidx(e, lane);
        b0[e] = Ks[buf][ln][kc];
        b1[e] = Ks[buf][16 + ln][kc];
      }
      s0 = __builtin_amdgcn_wmma_f32_16x16x32_bf16(false, qf[t], false, b0, (short)0, s0, false, false);
      s1 = __builtin_amdgcn_wmma_f32_16x16x32_bf16(false, qf[t], false, b1, (short)0, s1, false, false);
    }

    // online softmax; C-layout row m = r + 8*half, col = ln
    #pragma unroll
    for (int r = 0; r < 8; ++r) {
      int m    = r + (half << 3);
      int qpos = qrow0 + m;
      float e0 = s0[r] * scale;
      float e1 = s1[r] * scale;
      if (kvbase + ln > qpos)      e0 = -1e30f;
      if (kvbase + 16 + ln > qpos) e1 = -1e30f;
      float rmax = fmaxf(e0, e1);
      #pragma unroll
      for (int off = 1; off < 16; off <<= 1) rmax = fmaxf(rmax, __shfl_xor(rmax, off, 32));
      float mnew  = fmaxf(mrun[r], rmax);
      float alpha = __expf(mrun[r] - mnew);
      mrun[r] = mnew;
      float p0 = __expf(e0 - mnew);
      float p1 = __expf(e1 - mnew);
      float rs = p0 + p1;
      #pragma unroll
      for (int off = 1; off < 16; off <<= 1) rs += __shfl_xor(rs, off, 32);
      lsum[r] = lsum[r] * alpha + rs;
      #pragma unroll
      for (int g = 0; g < 8; ++g) o[g][r] *= alpha;
      Ps[wave][m * 32 + ln]      = (__bf16)p0;
      Ps[wave][m * 32 + 16 + ln] = (__bf16)p1;
    }

    // reload P in A-layout; o += P x V
    v16bf pa;
    #pragma unroll
    for (int e = 0; e < 16; ++e) pa[e] = Ps[wave][ln * 32 + a_kidx(e, lane)];
    #pragma unroll
    for (int g = 0; g < 8; ++g) {
      v16bf bv;
      #pragma unroll
      for (int e = 0; e < 16; ++e) bv[e] = Vs[buf][b_kidx(e, lane)][g * 16 + ln];
      o[g] = __builtin_amdgcn_wmma_f32_16x16x32_bf16(false, pa, false, bv, (short)0, o[g], false, false);
    }

    if (j + 1 < nj) { wait_async0(); __syncthreads(); }
    buf ^= 1;
  }

  // normalize, write attended bf16 in [B, S, H*128]
  #pragma unroll
  for (int r = 0; r < 8; ++r) {
    float linv = 1.0f / lsum[r];
    int row = qrow0 + r + (half << 3);
    __bf16* dst = att + ((size_t)b * S + row) * D_MODEL + h * D_HEAD + ln;
    #pragma unroll
    for (int g = 0; g < 8; ++g) dst[g * 16] = (__bf16)(o[g][r] * linv);
  }
}

// ---------------------------------------------------------------------------
extern "C" void kernel_launch(void* const* d_in, const int* in_sizes, int n_in,
                              void* d_out, int out_size, void* d_ws, size_t ws_size,
                              hipStream_t stream) {
  const float* qkv = (const float*)d_in[0];
  const float* Wq  = (const float*)d_in[1];
  const float* Wk  = (const float*)d_in[2];
  const float* Wv  = (const float*)d_in[3];
  const float* Wo  = (const float*)d_in[4];
  const float* qg  = (const float*)d_in[5];
  const float* kg  = (const float*)d_in[6];

  char* ws = (char*)d_ws;
  __bf16* xb   = (__bf16*)(ws);                  // 16.8 MB  qkv bf16
  __bf16* wqb  = (__bf16*)(ws + 16777216);       //  8.4 MB
  __bf16* wkb  = (__bf16*)(ws + 25165824);       //  2.1 MB
  __bf16* wvb  = (__bf16*)(ws + 27262976);       //  2.1 MB
  __bf16* wob  = (__bf16*)(ws + 29360128);       //  8.4 MB
  float*  qf   = (float*)(ws + 37748736);        // 33.5 MB  (region reused by attb)
  float*  kf   = (float*)(ws + 71303168);        //  8.4 MB
  float*  vf   = (float*)(ws + 79691776);        //  8.4 MB
  __bf16* qb   = (__bf16*)(ws + 88080384);       // 16.8 MB  [B,16,S,128]
  __bf16* kb   = (__bf16*)(ws + 104857600);      //  4.2 MB  [B,4,S,128]
  __bf16* vb   = (__bf16*)(ws + 109051904);      //  4.2 MB
  __bf16* attb = (__bf16*)(ws + 37748736);       // 16.8 MB  (qf consumed first)

  const int M = BATCH * SEQ;                     // 4096
  const int NQ = D_MODEL, NKV = NUM_KV_HEADS * D_HEAD;  // 2048, 512

  // one-time f32 -> bf16 precision shift of activations + weights
  cvt_f32_bf16<<<dim3(M * D_MODEL / 2048), 256, 0, stream>>>(qkv, xb, M * D_MODEL);
  cvt_f32_bf16<<<dim3(NQ * D_MODEL / 2048), 256, 0, stream>>>(Wq, wqb, NQ * D_MODEL);
  cvt_f32_bf16<<<dim3(NKV * D_MODEL / 2048), 256, 0, stream>>>(Wk, wkb, NKV * D_MODEL);
  cvt_f32_bf16<<<dim3(NKV * D_MODEL / 2048), 256, 0, stream>>>(Wv, wvb, NKV * D_MODEL);
  cvt_f32_bf16<<<dim3(NQ * D_MODEL / 2048), 256, 0, stream>>>(Wo, wob, NQ * D_MODEL);

  // projections: Y = X * W^T  (block tile 128x128)
  gemm_bf16<<<dim3(NQ / 128,  M / 128), 128, 0, stream>>>(xb, wqb, qf, M, NQ,  D_MODEL);
  gemm_bf16<<<dim3(NKV / 128, M / 128), 128, 0, stream>>>(xb, wkb, kf, M, NKV, D_MODEL);
  gemm_bf16<<<dim3(NKV / 128, M / 128), 128, 0, stream>>>(xb, wvb, vf, M, NKV, D_MODEL);

  // RMSNorm + RoPE + head-major transpose (v: transpose/convert only)
  norm_rope_kernel<<<dim3(M, 4), 128, 0, stream>>>(qf, qg, qb, NUM_HEADS,    SEQ, 1);
  norm_rope_kernel<<<dim3(M, 1), 128, 0, stream>>>(kf, kg, kb, NUM_KV_HEADS, SEQ, 1);
  norm_rope_kernel<<<dim3(M, 1), 128, 0, stream>>>(vf, qg, vb, NUM_KV_HEADS, SEQ, 0);

  // causal GQA flash attention -> bf16 attended
  flash_attn_kernel<<<dim3(SEQ / 64, NUM_HEADS, BATCH), 128, 0, stream>>>(qb, kb, vb, attb, SEQ);

  // output projection
  gemm_bf16<<<dim3(NQ / 128, M / 128), 128, 0, stream>>>(attb, wob, (float*)d_out, M, NQ, D_MODEL);
}
